// Encoder_20890720928284
// MI455X (gfx1250) — compile-verified, matrix operand order checked by reading, not run on hardware
//
#include <hip/hip_runtime.h>
#include <hip/hip_bf16.h>

typedef __attribute__((ext_vector_type(16))) __bf16 v16bf;
typedef __attribute__((ext_vector_type(8)))  float  v8f;

#define BATCH 64
#define H0    480
#define W0    80
#define C0    8
#define C1    16
#define H2    160
#define W2    40
#define C2    16
#define C3    32
#define TT    40
#define WP    20
#define IG    640   // WP*C3
#define HID   256
#define G3    768   // 3*HID
#define HW0   (H0*W0)    // 38400
#define HW2   (H2*W2)    // 6400

// ---------------- workspace layout (bytes) ----------------
#define MB1 ((size_t)1024*1024)
#define OFS_A1    ((size_t)0)              // conv1 out NHWC bf16, 78.6 MB
#define OFS_A4    ((size_t)0)              // conv3 out NHWC bf16, 26.2 MB (a1 dead)
#define OFS_XB    (32*MB1)                 // GRU input bf16 (B,T,640), 3.3 MB
#define OFS_GXF   (40*MB1)                 // 7.9 MB
#define OFS_GXB   (56*MB1)                 // 7.9 MB
#define OFS_A0    (80*MB1)                 // conv0 out NHWC bf16, 39.3 MB
#define OFS_A2    (80*MB1)                 // pool1 out bf16, 13.1 MB (a0 dead)
#define OFS_A3    (96*MB1)                 // conv2 out bf16, 13.1 MB
#define OFS_WB1   (120*MB1)                // packed conv1 w, 16x96 bf16
#define OFS_WB2   (OFS_WB1 + 0x2000)       // packed conv2 w, 16x160 bf16
#define OFS_WB3   (OFS_WB2 + 0x2000)       // packed conv3 w, 32x160 bf16
#define OFS_WIHF  (OFS_WB3 + 0x4000)
#define OFS_WIHB  (OFS_WIHF + 0x100000)
#define OFS_WHHF  (OFS_WIHB + 0x100000)
#define OFS_WHHB  (OFS_WHHF + 0x80000)
#define OFS_HF32  (OFS_WHHB + 0x80000)     // [dir][parity][64*256] f32
#define OFS_HBF   (OFS_HF32 + 0x40000)     // [dir][parity][64*256] bf16

// ============ WMMA fragment helpers (ISA 16-bit A/B layout) ============
// lanes 0-15 : VGPR0-3 = K k0..k0+7,  VGPR4-7 = K k0+16..k0+23
// lanes 16-31: VGPR0-3 = K k0+8..+15, VGPR4-7 = K k0+24..+31
__device__ __forceinline__ v16bf load_frag(const __bf16* __restrict__ p0, int row,
                                           int ld, int k0, int lane) {
  const __bf16* p = p0 + (size_t)(row + (lane & 15)) * ld + k0 + ((lane >> 4) << 3);
  struct U { uint4 a, b; } u;
  u.a = *(const uint4*)p;
  u.b = *(const uint4*)(p + 16);
  return __builtin_bit_cast(v16bf, u);
}

__device__ __forceinline__ uint4 ldg16_guard(const __bf16* __restrict__ base,
                                             int off, bool v) {
  uint4 r = *(const uint4*)(base + (v ? off : 0));
  if (!v) { r.x = 0u; r.y = 0u; r.z = 0u; r.w = 0u; }
  return r;
}

__device__ __forceinline__ v16bf frag_from(uint4 lo, uint4 hi) {
  struct U { uint4 a, b; } u; u.a = lo; u.b = hi;
  return __builtin_bit_cast(v16bf, u);
}

// ========= weight prepack: Wb[n][K], K = tap*Cin + ci, zero padded =========
__global__ void k_pack_w8(const float* __restrict__ w, __bf16* __restrict__ o) {
  int idx = blockIdx.x * blockDim.x + threadIdx.x;   // 16*96
  if (idx >= 16 * 96) return;
  int k = idx % 96, n = idx / 96;
  int tap = k / 8, ci = k % 8;
  float v = (tap < 9) ? w[(n * 8 + ci) * 9 + tap] : 0.0f;
  o[idx] = (__bf16)v;
}

__global__ void k_pack_w16(const float* __restrict__ w, __bf16* __restrict__ o, int cout) {
  int idx = blockIdx.x * blockDim.x + threadIdx.x;   // cout*160
  if (idx >= cout * 160) return;
  int k = idx % 160, n = idx / 160;
  int tap = k / 16, ci = k % 16;
  float v = (tap < 9) ? w[(n * 16 + ci) * 9 + tap] : 0.0f;
  o[idx] = (__bf16)v;
}

// ========= conv0: 1->8, 3x3 pad1, relu -> NHWC bf16 (VALU, 0.35 GFLOP) =========
__global__ void k_conv0_nhwc(const float* __restrict__ in, const float* __restrict__ w,
                             const float* __restrict__ bias, __bf16* __restrict__ out) {
  int idx = blockIdx.x * blockDim.x + threadIdx.x;   // B*HW0*C0
  int c = idx & 7; int P = idx >> 3;
  int b = P / HW0; int p = P - b * HW0;
  int h = p / W0, wv = p - h * W0;
  const float* wp = w + c * 9;
  const float* ip = in + (size_t)b * HW0;
  float acc = bias[c];
  #pragma unroll
  for (int dh = -1; dh <= 1; ++dh) {
    int hh = h + dh; if ((unsigned)hh >= H0) continue;
    #pragma unroll
    for (int dw = -1; dw <= 1; ++dw) {
      int ww = wv + dw; if ((unsigned)ww >= W0) continue;
      acc += ip[hh * W0 + ww] * wp[(dh + 1) * 3 + (dw + 1)];
    }
  }
  out[idx] = (__bf16)fmaxf(acc, 0.0f);
}

// ===== conv1: 8->16 via implicit-GEMM WMMA. K chunk = 4 taps x 8 ch =====
__global__ void k_conv1_wmma(const __bf16* __restrict__ in, const __bf16* __restrict__ wb,
                             const float* __restrict__ bias, __bf16* __restrict__ out) {
  int wid  = blockIdx.x * (blockDim.x >> 5) + (threadIdx.x >> 5);
  int lane = threadIdx.x & 31;
  int hsel = lane >> 4;
  int Pl = wid * 16 + (lane & 15);
  int b = Pl / HW0; int p = Pl - b * HW0;
  int h = p / W0, wv = p - h * W0;
  int bbase = b * HW0;
  v8f acc = {};
  #pragma unroll
  for (int kc = 0; kc < 3; ++kc) {
    int ta = 4 * kc + hsel;        // lanes<16: taps 4kc,4kc+2 ; lanes>=16: 4kc+1,4kc+3
    int tb = ta + 2;
    int ha = h + ta / 3 - 1, wa = wv + ta % 3 - 1;
    int hb = h + tb / 3 - 1, wbv = wv + tb % 3 - 1;
    bool va = (ta < 9) && ((unsigned)ha < H0) && ((unsigned)wa < W0);
    bool vb = (tb < 9) && ((unsigned)hb < H0) && ((unsigned)wbv < W0);
    int oa = (bbase + ha * W0 + wa) * C0;
    int ob = (bbase + hb * W0 + wbv) * C0;
    uint4 lo = ldg16_guard(in, oa, va);
    uint4 hi = ldg16_guard(in, ob, vb);
    v16bf a  = frag_from(lo, hi);
    v16bf bm = load_frag(wb, 0, 96, kc * 32, lane);
    acc = __builtin_amdgcn_wmma_f32_16x16x32_bf16(false, a, false, bm, (short)0,
                                                  acc, false, false);
  }
  int n = lane & 15;
  float bv = bias[n];
  int mbase = wid * 16 + 8 * hsel;
  #pragma unroll
  for (int r = 0; r < 8; ++r)
    out[(size_t)(mbase + r) * C1 + n] = (__bf16)fmaxf(acc[r] + bv, 0.0f);
}

// ===== convN: 16->Cout via implicit-GEMM WMMA. K chunk = 2 taps x 16 ch =====
__global__ void k_conv16_wmma(const __bf16* __restrict__ in, const __bf16* __restrict__ wb,
                              const float* __restrict__ bias, __bf16* __restrict__ out,
                              int H, int W, int nt /*Cout/16*/) {
  int wid  = blockIdx.x * (blockDim.x >> 5) + (threadIdx.x >> 5);
  int lane = threadIdx.x & 31;
  int pt = wid / nt, ntile = wid - pt * nt;
  int hsel = lane >> 4;
  int HWl = H * W;
  int Pl = pt * 16 + (lane & 15);
  int b = Pl / HWl; int p = Pl - b * HWl;
  int h = p / W, wv = p - h * W;
  int bbase = b * HWl;
  int cout = nt * 16;
  v8f acc = {};
  #pragma unroll
  for (int kc = 0; kc < 5; ++kc) {
    int t1 = 2 * kc, t2 = 2 * kc + 1;
    int h1 = h + t1 / 3 - 1, w1 = wv + t1 % 3 - 1;
    int h2 = h + t2 / 3 - 1, w2 = wv + t2 % 3 - 1;
    bool v1 = ((unsigned)h1 < (unsigned)H) && ((unsigned)w1 < (unsigned)W);
    bool v2 = (t2 < 9) && ((unsigned)h2 < (unsigned)H) && ((unsigned)w2 < (unsigned)W);
    int o1 = (bbase + h1 * W + w1) * 16 + hsel * 8;
    int o2 = (bbase + h2 * W + w2) * 16 + hsel * 8;
    uint4 lo = ldg16_guard(in, o1, v1);
    uint4 hi = ldg16_guard(in, o2, v2);
    v16bf a  = frag_from(lo, hi);
    v16bf bm = load_frag(wb, ntile * 16, 160, kc * 32, lane);
    acc = __builtin_amdgcn_wmma_f32_16x16x32_bf16(false, a, false, bm, (short)0,
                                                  acc, false, false);
  }
  int n = ntile * 16 + (lane & 15);
  float bv = bias[n];
  int mbase = pt * 16 + 8 * hsel;
  #pragma unroll
  for (int r = 0; r < 8; ++r)
    out[(size_t)(mbase + r) * cout + n] = (__bf16)fmaxf(acc[r] + bv, 0.0f);
}

// ================= maxpool(3,2) NHWC bf16 =================
__global__ void k_pool1(const __bf16* __restrict__ a1, __bf16* __restrict__ a2) {
  int idx = blockIdx.x * blockDim.x + threadIdx.x;   // B*H2*W2*C1
  int c = idx & 15; int t = idx >> 4;
  int pw = t % W2; t /= W2;
  int ph = t % H2; int b = t / H2;
  float mx = -1e30f;
  #pragma unroll
  for (int i = 0; i < 3; ++i)
    #pragma unroll
    for (int j = 0; j < 2; ++j) {
      float v = (float)a1[((size_t)(b * H0 + ph * 3 + i) * W0 + pw * 2 + j) * C1 + c];
      mx = fmaxf(mx, v);
    }
  a2[idx] = (__bf16)mx;
}

// ===== maxpool(4,2) NHWC bf16 -> GRU input (B,T,640) =====
__global__ void k_pool2(const __bf16* __restrict__ a4, __bf16* __restrict__ xb) {
  int idx = blockIdx.x * blockDim.x + threadIdx.x;   // B*TT*WP*C3
  int c = idx & 31; int t = idx >> 5;
  int pw = t % WP; t /= WP;
  int tt = t % TT; int b = t / TT;
  float mx = -1e30f;
  #pragma unroll
  for (int i = 0; i < 4; ++i)
    #pragma unroll
    for (int j = 0; j < 2; ++j) {
      float v = (float)a4[((size_t)(b * H2 + tt * 4 + i) * W2 + pw * 2 + j) * C3 + c];
      mx = fmaxf(mx, v);
    }
  xb[idx] = (__bf16)mx;
}

// ================= fp32 -> bf16 =================
__global__ void k_f2bf(const float* __restrict__ s, __bf16* __restrict__ d, int n) {
  int i = blockIdx.x * blockDim.x + threadIdx.x;
  if (i < n) d[i] = (__bf16)s[i];
}

// ================= init hidden (parity 0 buffers, fp32 + bf16) =================
__global__ void k_hinit(const float* __restrict__ hidden, float* __restrict__ hf,
                        __bf16* __restrict__ hb) {
  int i = blockIdx.x * blockDim.x + threadIdx.x;     // 2*64*256
  int dir = i / (BATCH * HID);
  int r   = i % (BATCH * HID);
  float v = hidden[i];
  hf[(size_t)(dir * 2) * BATCH * HID + r] = v;
  hb[(size_t)(dir * 2) * BATCH * HID + r] = (__bf16)v;
}

// ====== gx = X(2560x640) * W_ih^T(640x768) + b_ih, both directions ======
__global__ void k_gx_gemm(const __bf16* __restrict__ xb,
                          const __bf16* __restrict__ wihf, const __bf16* __restrict__ wihb,
                          const float* __restrict__ bihf, const float* __restrict__ bihb,
                          float* __restrict__ gxf, float* __restrict__ gxb) {
  int wid  = blockIdx.x * (blockDim.x >> 5) + (threadIdx.x >> 5);
  int lane = threadIdx.x & 31;
  const int MT = (TT * BATCH) / 16;   // 160
  const int NT = G3 / 16;             // 48
  int dir = wid / (MT * NT);
  int rem = wid % (MT * NT);
  int tm = rem % MT, tn = rem / MT;
  const __bf16* wih = dir ? wihb : wihf;
  const float*  bih = dir ? bihb : bihf;
  float*        gx  = dir ? gxb  : gxf;
  int mb = tm * 16, nb = tn * 16;
  v8f acc = {};
  #pragma unroll 4
  for (int k0 = 0; k0 < IG; k0 += 32) {
    v16bf a = load_frag(xb,  mb, IG, k0, lane);
    v16bf b = load_frag(wih, nb, IG, k0, lane);   // B[k][n] = wih[n][k]
    acc = __builtin_amdgcn_wmma_f32_16x16x32_bf16(false, a, false, b, (short)0,
                                                  acc, false, false);
  }
  int n   = nb + (lane & 15);
  float bv = bih[n];
  int mlo = mb + ((lane >> 4) << 3);
  #pragma unroll
  for (int r = 0; r < 8; ++r)
    gx[(size_t)(mlo + r) * G3 + n] = acc[r] + bv;
}

// ====== one GRU step, both directions fused (gh GEMM + gates + output) ======
__global__ void k_gru_step(const float* __restrict__ gxf, const float* __restrict__ gxb,
                           const __bf16* __restrict__ whhf, const __bf16* __restrict__ whhb,
                           const float* __restrict__ bhhf, const float* __restrict__ bhhb,
                           float* __restrict__ hf32, __bf16* __restrict__ hbf,
                           float* __restrict__ out, int s) {
  int wid  = blockIdx.x * (blockDim.x >> 5) + (threadIdx.x >> 5);  // 0..127
  int lane = threadIdx.x & 31;
  int dir  = wid >> 6;
  int tile = wid & 63;
  int tm = tile & 3, tn = tile >> 2;       // M=64 -> 4 tiles, N=256 -> 16 tiles
  int mb = tm * 16, nb = tn * 16;
  int cur = s & 1, nxt = cur ^ 1;
  const float*  gx   = dir ? gxb  : gxf;
  const __bf16* whh  = dir ? whhb : whhf;
  const float*  bhh  = dir ? bhhb : bhhf;
  const __bf16* hin  = hbf  + (size_t)(dir * 2 + cur) * BATCH * HID;
  const float*  hinf = hf32 + (size_t)(dir * 2 + cur) * BATCH * HID;
  float*  houtf = hf32 + (size_t)(dir * 2 + nxt) * BATCH * HID;
  __bf16* houtb = hbf  + (size_t)(dir * 2 + nxt) * BATCH * HID;
  int t = dir ? (TT - 1 - s) : s;

  v8f ar = {}, az = {}, an = {};
  #pragma unroll
  for (int k0 = 0; k0 < HID; k0 += 32) {
    v16bf a  = load_frag(hin, mb, HID, k0, lane);
    v16bf br = load_frag(whh, 0 * HID + nb, HID, k0, lane);
    v16bf bz = load_frag(whh, 1 * HID + nb, HID, k0, lane);
    v16bf bn = load_frag(whh, 2 * HID + nb, HID, k0, lane);
    ar = __builtin_amdgcn_wmma_f32_16x16x32_bf16(false, a, false, br, (short)0, ar, false, false);
    az = __builtin_amdgcn_wmma_f32_16x16x32_bf16(false, a, false, bz, (short)0, az, false, false);
    an = __builtin_amdgcn_wmma_f32_16x16x32_bf16(false, a, false, bn, (short)0, an, false, false);
  }
  int nh = nb + (lane & 15);
  float brc = bhh[nh], bzc = bhh[HID + nh], bnc = bhh[2 * HID + nh];
  int mlo = mb + ((lane >> 4) << 3);
  #pragma unroll
  for (int r = 0; r < 8; ++r) {
    int m = mlo + r;
    size_t gb = (size_t)(t * BATCH + m) * G3;
    float xr = gx[gb + nh];
    float xz = gx[gb + HID + nh];
    float xn = gx[gb + 2 * HID + nh];
    float hr = ar[r] + brc;
    float hz = az[r] + bzc;
    float hn = an[r] + bnc;
    float rg = 1.0f / (1.0f + expf(-(xr + hr)));
    float zg = 1.0f / (1.0f + expf(-(xz + hz)));
    float ng = tanhf(xn + rg * hn);
    float hp = hinf[m * HID + nh];
    float hnew = (1.0f - zg) * ng + zg * hp;
    houtf[m * HID + nh] = hnew;
    houtb[m * HID + nh] = (__bf16)hnew;
    out[((size_t)m * TT + t) * (2 * HID) + dir * HID + nh] = hnew;
  }
}

// ================= final hidden copy (parity 0 after 40 steps) =================
__global__ void k_hfinal(const float* __restrict__ hf32, float* __restrict__ out) {
  int i = blockIdx.x * blockDim.x + threadIdx.x;     // 2*64*256
  int dir = i / (BATCH * HID);
  int r   = i % (BATCH * HID);
  out[(size_t)BATCH * TT * 2 * HID + i] = hf32[(size_t)(dir * 2) * BATCH * HID + r];
}

extern "C" void kernel_launch(void* const* d_in, const int* in_sizes, int n_in,
                              void* d_out, int out_size, void* d_ws, size_t ws_size,
                              hipStream_t stream) {
  const float* in_img = (const float*)d_in[0];
  const float* hidden = (const float*)d_in[1];
  const float* c0w = (const float*)d_in[2];  const float* c0b = (const float*)d_in[3];
  const float* c1w = (const float*)d_in[4];  const float* c1b = (const float*)d_in[5];
  const float* c2w = (const float*)d_in[6];  const float* c2b = (const float*)d_in[7];
  const float* c3w = (const float*)d_in[8];  const float* c3b = (const float*)d_in[9];
  const float* wihf = (const float*)d_in[10]; const float* whhf = (const float*)d_in[11];
  const float* bihf = (const float*)d_in[12]; const float* bhhf = (const float*)d_in[13];
  const float* wihb = (const float*)d_in[14]; const float* whhb = (const float*)d_in[15];
  const float* bihb = (const float*)d_in[16]; const float* bhhb = (const float*)d_in[17];
  float* out = (float*)d_out;
  char* ws = (char*)d_ws;

  __bf16* a0  = (__bf16*)(ws + OFS_A0);
  __bf16* a1  = (__bf16*)(ws + OFS_A1);
  __bf16* a2  = (__bf16*)(ws + OFS_A2);
  __bf16* a3  = (__bf16*)(ws + OFS_A3);
  __bf16* a4  = (__bf16*)(ws + OFS_A4);
  __bf16* xb  = (__bf16*)(ws + OFS_XB);
  __bf16* wb1 = (__bf16*)(ws + OFS_WB1);
  __bf16* wb2 = (__bf16*)(ws + OFS_WB2);
  __bf16* wb3 = (__bf16*)(ws + OFS_WB3);
  __bf16* wihf_b = (__bf16*)(ws + OFS_WIHF);
  __bf16* wihb_b = (__bf16*)(ws + OFS_WIHB);
  __bf16* whhf_b = (__bf16*)(ws + OFS_WHHF);
  __bf16* whhb_b = (__bf16*)(ws + OFS_WHHB);
  float*  gxf  = (float*)(ws + OFS_GXF);
  float*  gxb  = (float*)(ws + OFS_GXB);
  float*  hf32 = (float*)(ws + OFS_HF32);
  __bf16* hbf  = (__bf16*)(ws + OFS_HBF);

  // weight prepack (tiny)
  k_pack_w8 <<<(16 * 96 + 255) / 256, 256, 0, stream>>>(c1w, wb1);
  k_pack_w16<<<(16 * 160 + 255) / 256, 256, 0, stream>>>(c2w, wb2, 16);
  k_pack_w16<<<(32 * 160 + 255) / 256, 256, 0, stream>>>(c3w, wb3, 32);
  k_f2bf<<<(G3 * IG  + 255) / 256, 256, 0, stream>>>(wihf, wihf_b, G3 * IG);
  k_f2bf<<<(G3 * IG  + 255) / 256, 256, 0, stream>>>(wihb, wihb_b, G3 * IG);
  k_f2bf<<<(G3 * HID + 255) / 256, 256, 0, stream>>>(whhf, whhf_b, G3 * HID);
  k_f2bf<<<(G3 * HID + 255) / 256, 256, 0, stream>>>(whhb, whhb_b, G3 * HID);
  k_hinit<<<(2 * BATCH * HID) / 256, 256, 0, stream>>>(hidden, hf32, hbf);

  // conv stack: conv0 VALU->NHWC bf16, conv1/2/3 implicit-GEMM WMMA
  k_conv0_nhwc<<<(BATCH * HW0 * C0) / 256, 256, 0, stream>>>(in_img, c0w, c0b, a0);
  k_conv1_wmma<<<(BATCH * HW0 / 16) / 8, 256, 0, stream>>>(a0, wb1, c1b, a1);     // 153600 waves
  k_pool1<<<(BATCH * HW2 * C1) / 256, 256, 0, stream>>>(a1, a2);
  k_conv16_wmma<<<(BATCH * HW2 / 16) * 1 / 8, 256, 0, stream>>>(a2, wb2, c2b, a3, H2, W2, 1);
  k_conv16_wmma<<<(BATCH * HW2 / 16) * 2 / 8, 256, 0, stream>>>(a3, wb3, c3b, a4, H2, W2, 2);
  k_pool2<<<(BATCH * TT * WP * C3) / 256, 256, 0, stream>>>(a4, xb);

  // big input GEMM for both directions: 2 * 160 * 48 = 15360 waves, 8 waves/block
  k_gx_gemm<<<15360 / 8, 256, 0, stream>>>(xb, wihf_b, wihb_b, bihf, bihb, gxf, gxb);

  // sequential GRU scan, forward+backward fused per step: 128 waves = 16 blocks
  for (int s = 0; s < TT; ++s) {
    k_gru_step<<<16, 256, 0, stream>>>(gxf, gxb, whhf_b, whhb_b, bhhf, bhhb,
                                       hf32, hbf, out, s);
  }

  // final hidden state
  k_hfinal<<<(2 * BATCH * HID) / 256, 256, 0, stream>>>(hf32, out);
}
